// PointTransformerLayer_22909355556975
// MI455X (gfx1250) — matrix operand order repaired and checked by reading, not compiled, not run on hardware
//
#include <hip/hip_runtime.h>
#include <hip/hip_bf16.h>

// ---------------- problem constants ----------------
constexpr int NB  = 4;     // batch
constexpr int NP  = 4096;  // points
constexpr int CH  = 64;    // channels
constexpr int KNN = 16;    // neighbors
constexpr float EPS = 1e-5f;

// transposed GEMM-operand pitch: 72 halfs = 144 B (16B aligned, 36-dword lane
// stride -> 16 lanes x 4 dwords cover all 64 LDS banks conflict-free)
constexpr int XP = 72;

typedef __attribute__((ext_vector_type(16))) _Float16 v16h;
typedef __attribute__((ext_vector_type(8)))  _Float16 v8h;
typedef __attribute__((ext_vector_type(8)))  float    v8f;

__device__ __forceinline__ v16h cat8(v8h a, v8h b) {
  return __builtin_shufflevector(a, b, 0, 1, 2, 3, 4, 5, 6, 7,
                                 8, 9, 10, 11, 12, 13, 14, 15);
}

// ---------------------------------------------------------------------------
// 64x64 @ 64x256 GEMM tile via V_WMMA_F32_16X16X32_F16, all-vector LDS access.
// W : [64][64] f16, row = out channel, col = in channel (A rows contiguous)
// XT: [256][XP] f16, TRANSPOSED: row = column, col = in channel (B K contiguous)
// Fragment layouts per CDNA5 ISA 7.12.2 (wave32):
//   A 16x32 f16 : M = lane%16 ; h 0..7 -> K=k0+aoff+h ; h 8..15 -> K=k0+aoff+16+(h-8)
//   B 32x16 f16 : N = lane%16 ; h 0..7 -> K=k0+boff+h ; h 8..15 -> K=k0+boff+16+(h-8)
//   D 16x16 f32 : N = lane%16 ; M(r) = mt*16 + r + 8*(lane>=16)
// ---------------------------------------------------------------------------
__device__ __forceinline__ v8f wmma_gemm64(const _Float16 (*W)[CH],
                                           const _Float16 (*XT)[XP],
                                           int mt, int nt, int lane) {
  v8f acc = {};
  const int row  = mt * 16 + (lane & 15);
  const int col  = nt * 16 + (lane & 15);
  const int aoff = (lane >= 16) ? 8 : 0;
  const int boff = (lane >= 16) ? 8 : 0;
#pragma unroll
  for (int ks = 0; ks < 2; ++ks) {
    const int k0 = ks * 32;
    const v8h a0 = *(const v8h*)&W[row][k0 + aoff];
    const v8h a1 = *(const v8h*)&W[row][k0 + aoff + 16];
    const v8h b0 = *(const v8h*)&XT[col][k0 + boff];
    const v8h b1 = *(const v8h*)&XT[col][k0 + boff + 16];
    acc = __builtin_amdgcn_wmma_f32_16x16x32_f16(false, cat8(a0, a1),
                                                 false, cat8(b0, b1),
                                                 (short)0, acc, false, false);
  }
  return acc;
}

// stage a 64x64 f32 weight matrix into LDS as f16 (vector stores):
// each thread converts 16 consecutive elements -> two ds_store_b128.
__device__ __forceinline__ void load_w64(_Float16 (*sW)[CH],
                                         const float* __restrict__ g, int tid) {
  const int base = tid * 16;  // 256 threads * 16 = 4096
  v8h p0, p1;
#pragma unroll
  for (int u = 0; u < 8; ++u) p0[u] = (_Float16)g[base + u];
#pragma unroll
  for (int u = 0; u < 8; ++u) p1[u] = (_Float16)g[base + 8 + u];
  *(v8h*)&sW[base >> 6][base & 63] = p0;
  *(v8h*)&sW[base >> 6][(base & 63) + 8] = p1;
}

// ---------------------------------------------------------------------------
// Kernel 1: q/k/v 1x1 conv.  grid = NB*16, 256 threads (8 waves).
// One batch, 256 columns of x; 3 GEMMs W[64x64] @ x[64x256] (WMMA f16->f32).
// ---------------------------------------------------------------------------
__global__ __launch_bounds__(256) void qkv_kernel(
    const float* __restrict__ x,
    const float* __restrict__ Wq, const float* __restrict__ bq,
    const float* __restrict__ Wk, const float* __restrict__ bk,
    const float* __restrict__ Wv, const float* __restrict__ bv,
    float* __restrict__ qo, float* __restrict__ ko, float* __restrict__ vo) {
  __shared__ _Float16 sXT[256][XP];
  __shared__ _Float16 sW[CH][CH];
  const int tid  = threadIdx.x;
  const int lane = tid & 31;
  const int wave = tid >> 5;
  const int b    = blockIdx.x >> 4;
  const int col0 = (blockIdx.x & 15) * 256;

  // x tile -> transposed LDS (coalesced global reads, vector LDS stores)
#pragma unroll 2
  for (int c0 = 0; c0 < CH; c0 += 8) {
    v8h pk;
#pragma unroll
    for (int u = 0; u < 8; ++u)
      pk[u] = (_Float16)x[((size_t)b * CH + c0 + u) * NP + col0 + tid];
    *(v8h*)&sXT[tid][c0] = pk;
  }

  const float* Ws[3] = {Wq, Wk, Wv};
  const float* bs[3] = {bq, bk, bv};
  float*       os[3] = {qo, ko, vo};

#pragma unroll
  for (int w = 0; w < 3; ++w) {
    __syncthreads();
    load_w64(sW, Ws[w], tid);
    __syncthreads();
    v8f accs[8];
#pragma unroll
    for (int t = 0; t < 8; ++t) {
      const int tile = wave * 8 + t;
      accs[t] = wmma_gemm64(sW, sXT, tile >> 4, tile & 15, lane);
    }
#pragma unroll
    for (int t = 0; t < 8; ++t) {
      const int tile = wave * 8 + t;
      const int mt = tile >> 4, nt = tile & 15;
      const int colg = col0 + nt * 16 + (lane & 15);
      const int m0 = mt * 16 + ((lane >= 16) ? 8 : 0);
#pragma unroll
      for (int r = 0; r < 8; ++r)
        os[w][((size_t)b * CH + m0 + r) * NP + colg] = accs[t][r] + bs[w][m0 + r];
    }
  }
}

// ---------------------------------------------------------------------------
// Kernel 2: exact KNN (top-16 by squared distance; zero points -> +inf).
// grid = NB*16, 256 threads; one query/thread; candidates staged via LDS.
// ---------------------------------------------------------------------------
__global__ __launch_bounds__(256) void knn_kernel(const float* __restrict__ p,
                                                  int* __restrict__ idxbuf) {
  __shared__ float spx[256], spy[256], spz[256], sbad[256];
  const int tid = threadIdx.x;
  const int b   = blockIdx.x >> 4;
  const int n   = (blockIdx.x & 15) * 256 + tid;
  const float* pb = p + (size_t)b * NP * 3;
  const float px = pb[n * 3 + 0];
  const float py = pb[n * 3 + 1];
  const float pz = pb[n * 3 + 2];

  float bd[KNN];
  int   bi[KNN];
#pragma unroll
  for (int t = 0; t < KNN; ++t) { bd[t] = 3.0e38f; bi[t] = 0; }

  for (int cj = 0; cj < NP / 256; ++cj) {
    __syncthreads();
    const int gj = cj * 256 + tid;
    const float qx = pb[gj * 3 + 0];
    const float qy = pb[gj * 3 + 1];
    const float qz = pb[gj * 3 + 2];
    spx[tid] = qx; spy[tid] = qy; spz[tid] = qz;
    sbad[tid] = (qx == 0.f && qy == 0.f && qz == 0.f) ? 1.0e37f : 0.f;
    __syncthreads();
#pragma unroll 1
    for (int j = 0; j < 256; ++j) {
      const float dx = px - spx[j], dy = py - spy[j], dz = pz - spz[j];
      const float d2 = dx * dx + dy * dy + dz * dz + sbad[j];
      if (d2 < bd[KNN - 1]) {
        float dc = d2; int ic = cj * 256 + j;
#pragma unroll
        for (int t = 0; t < KNN; ++t) {
          if (dc < bd[t]) {
            const float td = bd[t]; const int ti = bi[t];
            bd[t] = dc; bi[t] = ic; dc = td; ic = ti;
          }
        }
      }
    }
  }
#pragma unroll
  for (int t = 0; t < KNN; ++t)
    idxbuf[((size_t)b * NP + n) * KNN + t] = bi[t];
}

// ---------------------------------------------------------------------------
// Kernel 3: fused gather + pos-enc MLP + attention MLP + softmax + aggregate.
// grid = NB*(NP/16), 256 threads; block = 16 queries x 16 neighbors = 256 cols.
// Three WMMA 64x64x256 GEMMs, transposed f16 LDS operands, register epilogues.
// ---------------------------------------------------------------------------
__global__ __launch_bounds__(256) void pt_fused_kernel(
    const float* __restrict__ p, const unsigned char* __restrict__ mask,
    const float* __restrict__ qbuf, const float* __restrict__ kbuf,
    const float* __restrict__ vbuf, const int* __restrict__ idxbuf,
    const float* __restrict__ pe_w1,
    const float* __restrict__ pe_g, const float* __restrict__ pe_b,
    const float* __restrict__ pe_m, const float* __restrict__ pe_v,
    const float* __restrict__ pe_w2, const float* __restrict__ pe_b2,
    const float* __restrict__ a0g, const float* __restrict__ a0b,
    const float* __restrict__ a0m, const float* __restrict__ a0v,
    const float* __restrict__ at_w1,
    const float* __restrict__ a1g, const float* __restrict__ a1b,
    const float* __restrict__ a1m, const float* __restrict__ a1v,
    const float* __restrict__ at_w2, const float* __restrict__ at_b2,
    float* __restrict__ y) {
  __shared__ _Float16 sBufT[256][XP];  // transposed ping buffer [col][chan]
  __shared__ _Float16 sNRT[256][XP];   // transposed n_r          [col][chan]
  __shared__ _Float16 sW[CH][CH];      // current weight
  __shared__ float    sQT[16][CH];     // q tile, [query][chan]
  __shared__ int      sIdx[256];
  __shared__ float    sRel[3][256];
  __shared__ float    sSc[3][CH], sBi[3][CH];  // pe_bn, at_bn0, at_bn1
  __shared__ float    sPw1[CH * 3];
  __shared__ float    sPb2[CH], sAb2[CH];

  const int tid  = threadIdx.x;
  const int lane = tid & 31;
  const int wave = tid >> 5;
  const int b    = blockIdx.x >> 8;            // NP/16 = 256 tiles per batch
  const int n0   = (blockIdx.x & 255) * 16;

  // ---- stage 0: indices, q tile, small params ----
  sIdx[tid] = idxbuf[((size_t)b * NP + n0 + (tid >> 4)) * KNN + (tid & 15)];
#pragma unroll
  for (int i = 0; i < 4; ++i) {
    const int e = i * 256 + tid;
    const int c = e >> 4, ql = e & 15;
    sQT[ql][c] = qbuf[((size_t)b * CH + c) * NP + n0 + ql];
  }
  if (tid < CH * 3) sPw1[tid] = pe_w1[tid];
  if (tid < CH) {
    const int c = tid;
    float s;
    s = pe_g[c] * rsqrtf(pe_v[c] + EPS); sSc[0][c] = s; sBi[0][c] = pe_b[c] - pe_m[c] * s;
    s = a0g[c] * rsqrtf(a0v[c] + EPS);   sSc[1][c] = s; sBi[1][c] = a0b[c] - a0m[c] * s;
    s = a1g[c] * rsqrtf(a1v[c] + EPS);   sSc[2][c] = s; sBi[2][c] = a1b[c] - a1m[c] * s;
    sPb2[c] = pe_b2[c];
    sAb2[c] = at_b2[c];
  }
  __syncthreads();

  // ---- stage 1: relative positions + stage pe_w2 ----
  {
    const int j  = sIdx[tid];
    const int nq = n0 + (tid >> 4);
#pragma unroll
    for (int d = 0; d < 3; ++d)
      sRel[d][tid] = p[((size_t)b * NP + nq) * 3 + d] - p[((size_t)b * NP + j) * 3 + d];
  }
  load_w64(sW, pe_w2, tid);
  __syncthreads();

  // ---- stage 2: h = relu(bn(pe_w1 @ rel)), K=3 conv in VALU, vector stores ----
  {
    const float r0 = sRel[0][tid], r1 = sRel[1][tid], r2 = sRel[2][tid];
#pragma unroll
    for (int c0 = 0; c0 < CH; c0 += 8) {
      v8h pk;
#pragma unroll
      for (int u = 0; u < 8; ++u) {
        const int c = c0 + u;
        float h = sPw1[c * 3 + 0] * r0 + sPw1[c * 3 + 1] * r1 + sPw1[c * 3 + 2] * r2;
        pk[u] = (_Float16)fmaxf(h * sSc[0][c] + sBi[0][c], 0.f);
      }
      *(v8h*)&sBufT[tid][c0] = pk;
    }
  }
  __syncthreads();

  v8f accs[8];

  // ---- stage 3: n_r = pe_w2 @ h + pe_b2 (WMMA) -> sNRT ----
#pragma unroll
  for (int t = 0; t < 8; ++t) {
    const int tile = wave * 8 + t;
    accs[t] = wmma_gemm64(sW, sBufT, tile >> 4, tile & 15, lane);
  }
  __syncthreads();
#pragma unroll
  for (int t = 0; t < 8; ++t) {
    const int tile = wave * 8 + t;
    const int mt = tile >> 4, nt = tile & 15;
    const int coll = nt * 16 + (lane & 15);
    const int m0 = mt * 16 + ((lane >= 16) ? 8 : 0);
    v8h pk;
#pragma unroll
    for (int r = 0; r < 8; ++r) pk[r] = (_Float16)(accs[t][r] + sPb2[m0 + r]);
    *(v8h*)&sNRT[coll][m0] = pk;
  }
  load_w64(sW, at_w1, tid);
  __syncthreads();

  // ---- stage 4: a0 = relu(bn0(q - n_k + n_r)) -> sBufT ----
  {
    const int j  = sIdx[tid];
    const int ql = tid >> 4;
#pragma unroll
    for (int c0 = 0; c0 < CH; c0 += 8) {
      const v8h nr = *(const v8h*)&sNRT[tid][c0];
      v8h pk;
#pragma unroll
      for (int u = 0; u < 8; ++u) {
        const int c = c0 + u;
        float a0 = sQT[ql][c] - kbuf[((size_t)b * CH + c) * NP + j] + (float)nr[u];
        pk[u] = (_Float16)fmaxf(a0 * sSc[1][c] + sBi[1][c], 0.f);
      }
      *(v8h*)&sBufT[tid][c0] = pk;
    }
  }
  __syncthreads();

  // ---- stage 5: a1 = relu(bn1(at_w1 @ a0)) (WMMA) -> sBufT ----
#pragma unroll
  for (int t = 0; t < 8; ++t) {
    const int tile = wave * 8 + t;
    accs[t] = wmma_gemm64(sW, sBufT, tile >> 4, tile & 15, lane);
  }
  __syncthreads();
#pragma unroll
  for (int t = 0; t < 8; ++t) {
    const int tile = wave * 8 + t;
    const int mt = tile >> 4, nt = tile & 15;
    const int coll = nt * 16 + (lane & 15);
    const int m0 = mt * 16 + ((lane >= 16) ? 8 : 0);
    v8h pk;
#pragma unroll
    for (int r = 0; r < 8; ++r)
      pk[r] = (_Float16)fmaxf(accs[t][r] * sSc[2][m0 + r] + sBi[2][m0 + r], 0.f);
    *(v8h*)&sBufT[coll][m0] = pk;
  }
  load_w64(sW, at_w2, tid);
  __syncthreads();

  // ---- stage 6: logits = at_w2 @ a1 + at_b2 (WMMA) -> sBufT ----
#pragma unroll
  for (int t = 0; t < 8; ++t) {
    const int tile = wave * 8 + t;
    accs[t] = wmma_gemm64(sW, sBufT, tile >> 4, tile & 15, lane);
  }
  __syncthreads();
#pragma unroll
  for (int t = 0; t < 8; ++t) {
    const int tile = wave * 8 + t;
    const int mt = tile >> 4, nt = tile & 15;
    const int coll = nt * 16 + (lane & 15);
    const int m0 = mt * 16 + ((lane >= 16) ? 8 : 0);
    v8h pk;
#pragma unroll
    for (int r = 0; r < 8; ++r) pk[r] = (_Float16)(accs[t][r] + sAb2[m0 + r]);
    *(v8h*)&sBufT[coll][m0] = pk;
  }
  __syncthreads();

  // ---- stage 7: softmax over K + weighted sum of (n_v + n_r) ----
#pragma unroll 1
  for (int i = 0; i < 4; ++i) {
    const int e  = i * 256 + tid;
    const int c  = e >> 4;
    const int ql = e & 15;
    const bool valid = mask[(size_t)b * NP + n0 + ql] != 0;
    float L[KNN];
#pragma unroll
    for (int kk = 0; kk < KNN; ++kk)
      L[kk] = valid ? (float)sBufT[ql * 16 + kk][c] : -1e10f;
    float mx = L[0];
#pragma unroll
    for (int kk = 1; kk < KNN; ++kk) mx = fmaxf(mx, L[kk]);
    float sum = 0.f;
#pragma unroll
    for (int kk = 0; kk < KNN; ++kk) { L[kk] = __expf(L[kk] - mx); sum += L[kk]; }
    const float inv = 1.f / sum;
    float acc = 0.f;
#pragma unroll
    for (int kk = 0; kk < KNN; ++kk) {
      const int j = sIdx[ql * 16 + kk];
      acc += L[kk] * inv *
             (vbuf[((size_t)b * CH + c) * NP + j] + (float)sNRT[ql * 16 + kk][c]);
    }
    y[((size_t)b * CH + c) * NP + n0 + ql] = acc;
  }
}

// ---------------------------------------------------------------------------
extern "C" void kernel_launch(void* const* d_in, const int* in_sizes, int n_in,
                              void* d_out, int out_size, void* d_ws, size_t ws_size,
                              hipStream_t stream) {
  (void)in_sizes; (void)n_in; (void)out_size; (void)ws_size;
  const float* p    = (const float*)d_in[0];
  const float* x    = (const float*)d_in[1];
  const unsigned char* mask = (const unsigned char*)d_in[2];
  const float* Wq = (const float*)d_in[3],  *bq = (const float*)d_in[4];
  const float* Wk = (const float*)d_in[5],  *bk = (const float*)d_in[6];
  const float* Wv = (const float*)d_in[7],  *bv = (const float*)d_in[8];
  const float* pe_w1 = (const float*)d_in[9];
  const float* pe_g  = (const float*)d_in[10], *pe_b = (const float*)d_in[11];
  const float* pe_m  = (const float*)d_in[12], *pe_v = (const float*)d_in[13];
  const float* pe_w2 = (const float*)d_in[14], *pe_b2 = (const float*)d_in[15];
  const float* a0g = (const float*)d_in[16], *a0b = (const float*)d_in[17];
  const float* a0m = (const float*)d_in[18], *a0v = (const float*)d_in[19];
  const float* at_w1 = (const float*)d_in[20];
  const float* a1g = (const float*)d_in[21], *a1b = (const float*)d_in[22];
  const float* a1m = (const float*)d_in[23], *a1v = (const float*)d_in[24];
  const float* at_w2 = (const float*)d_in[25], *at_b2 = (const float*)d_in[26];

  float* qbuf = (float*)d_ws;                         // [NB,CH,NP]
  float* kbuf = qbuf + (size_t)NB * CH * NP;
  float* vbuf = kbuf + (size_t)NB * CH * NP;
  int*   idxb = (int*)(vbuf + (size_t)NB * CH * NP);  // [NB,NP,KNN]

  qkv_kernel<<<NB * 16, 256, 0, stream>>>(x, Wq, bq, Wk, bk, Wv, bv,
                                          qbuf, kbuf, vbuf);
  knn_kernel<<<NB * 16, 256, 0, stream>>>(p, idxb);

  // output tuple: (p, y) concatenated flat
  hipMemcpyAsync(d_out, p, (size_t)NB * NP * 3 * sizeof(float),
                 hipMemcpyDeviceToDevice, stream);
  float* y = (float*)d_out + (size_t)NB * NP * 3;

  pt_fused_kernel<<<NB * (NP / 16), 256, 0, stream>>>(
      p, mask, qbuf, kbuf, vbuf, idxb,
      pe_w1, pe_g, pe_b, pe_m, pe_v, pe_w2, pe_b2,
      a0g, a0b, a0m, a0v, at_w1,
      a1g, a1b, a1m, a1v, at_w2, at_b2, y);
}
